// SelfAttention_6511170421201
// MI455X (gfx1250) — matrix operand order repaired and checked by reading, not compiled, not run on hardware
//
#include <hip/hip_runtime.h>

// ---------------------------------------------------------------------------
// Self-attention block for MI455X (gfx1250), bf16 WMMA compute, fp32 accum.
// B=2, T=2048, C=1024, H=16, d=64.
// ---------------------------------------------------------------------------

typedef __attribute__((ext_vector_type(16))) __bf16 v16bf;
typedef __attribute__((ext_vector_type(8)))  __bf16 v8bf;
typedef __attribute__((ext_vector_type(8)))  float  v8f;

__device__ __forceinline__ __bf16 f2bf(float f) { return (__bf16)f; }
__device__ __forceinline__ float  bf2f(__bf16 b) { return (float)b; }

__device__ __forceinline__ v16bf make16(v8bf lo, v8bf hi) {
  v16bf r;
#pragma unroll
  for (int i = 0; i < 8; ++i) { r[i] = lo[i]; r[i + 8] = hi[i]; }
  return r;
}
__device__ __forceinline__ v8f wmma_bf16(v16bf a, v16bf b, v8f c) {
  return __builtin_amdgcn_wmma_f32_16x16x32_bf16(false, a, false, b,
                                                 (short)0, c, false, false);
}

// ---------------------------------------------------------------------------
// Kernel 1/4: GEMM  out[M,N] = A[M,K] * B[N,K]^T + bias[N]
//   A: fp32 or bf16 (template), B: fp32 weights, out: bf16 or fp32.
// Block tile 128x128, K-step 32, 256 threads (8 waves), each wave 32x64.
// ---------------------------------------------------------------------------
#define LDS_STRIDE 40  // bf16 elements per LDS row (32 data + 8 pad)

template <bool A_BF16, bool OUT_BF16>
__global__ void __launch_bounds__(256)
wmma_gemm(const void* __restrict__ Araw, const float* __restrict__ Bm,
          const float* __restrict__ bias, void* __restrict__ Oraw,
          int M, int N, int K) {
  __shared__ __bf16 sA[128 * LDS_STRIDE];
  __shared__ __bf16 sB[128 * LDS_STRIDE];

  const int tid  = threadIdx.x;
  const int wave = tid >> 5;
  const int lane = tid & 31;
  const int m0 = blockIdx.y * 128;
  const int n0 = blockIdx.x * 128;
  const int wm = (wave >> 1) * 32;   // wave row offset in block tile
  const int wn = (wave & 1) * 64;    // wave col offset in block tile

  const int halfl = lane >> 4;       // 0/1 (lane group)
  const int l16   = lane & 15;

  v8f acc[2][4];
#pragma unroll
  for (int i = 0; i < 2; ++i)
#pragma unroll
    for (int j = 0; j < 4; ++j)
#pragma unroll
      for (int r = 0; r < 8; ++r) acc[i][j][r] = 0.0f;

  const int lrow  = tid >> 1;  // 0..127: tile row this thread stages
  const int lhalf = tid & 1;   // 0/1: which 16-wide K segment

  for (int k0 = 0; k0 < K; k0 += 32) {
    // ---- stage A tile (convert to bf16 if needed) ----
    if (A_BF16) {
      const __bf16* A = (const __bf16*)Araw;
      const __bf16* src = A + (size_t)(m0 + lrow) * K + k0 + lhalf * 16;
      if (k0 + 32 < K) __builtin_prefetch(src + 32, 0, 0);
      v8bf x0 = *(const v8bf*)src;
      v8bf x1 = *(const v8bf*)(src + 8);
      *(v8bf*)&sA[lrow * LDS_STRIDE + lhalf * 16]     = x0;
      *(v8bf*)&sA[lrow * LDS_STRIDE + lhalf * 16 + 8] = x1;
    } else {
      const float* A = (const float*)Araw;
      const float* src = A + (size_t)(m0 + lrow) * K + k0 + lhalf * 16;
      if (k0 + 32 < K) __builtin_prefetch(src + 32, 0, 0);
      float4 f0 = ((const float4*)src)[0];
      float4 f1 = ((const float4*)src)[1];
      float4 f2 = ((const float4*)src)[2];
      float4 f3 = ((const float4*)src)[3];
      v8bf p0, p1;
      p0[0] = f2bf(f0.x); p0[1] = f2bf(f0.y); p0[2] = f2bf(f0.z); p0[3] = f2bf(f0.w);
      p0[4] = f2bf(f1.x); p0[5] = f2bf(f1.y); p0[6] = f2bf(f1.z); p0[7] = f2bf(f1.w);
      p1[0] = f2bf(f2.x); p1[1] = f2bf(f2.y); p1[2] = f2bf(f2.z); p1[3] = f2bf(f2.w);
      p1[4] = f2bf(f3.x); p1[5] = f2bf(f3.y); p1[6] = f2bf(f3.z); p1[7] = f2bf(f3.w);
      *(v8bf*)&sA[lrow * LDS_STRIDE + lhalf * 16]     = p0;
      *(v8bf*)&sA[lrow * LDS_STRIDE + lhalf * 16 + 8] = p1;
    }
    // ---- stage B tile (always fp32 -> bf16) ----
    {
      const float* src = Bm + (size_t)(n0 + lrow) * K + k0 + lhalf * 16;
      if (k0 + 32 < K) __builtin_prefetch(src + 32, 0, 0);
      float4 f0 = ((const float4*)src)[0];
      float4 f1 = ((const float4*)src)[1];
      float4 f2 = ((const float4*)src)[2];
      float4 f3 = ((const float4*)src)[3];
      v8bf p0, p1;
      p0[0] = f2bf(f0.x); p0[1] = f2bf(f0.y); p0[2] = f2bf(f0.z); p0[3] = f2bf(f0.w);
      p0[4] = f2bf(f1.x); p0[5] = f2bf(f1.y); p0[6] = f2bf(f1.z); p0[7] = f2bf(f1.w);
      p1[0] = f2bf(f2.x); p1[1] = f2bf(f2.y); p1[2] = f2bf(f2.z); p1[3] = f2bf(f2.w);
      p1[4] = f2bf(f3.x); p1[5] = f2bf(f3.y); p1[6] = f2bf(f3.z); p1[7] = f2bf(f3.w);
      *(v8bf*)&sB[lrow * LDS_STRIDE + lhalf * 16]     = p0;
      *(v8bf*)&sB[lrow * LDS_STRIDE + lhalf * 16 + 8] = p1;
    }
    __syncthreads();

    // ---- fragments + 8 WMMAs ----
    v16bf afrag[2], bfrag[4];
#pragma unroll
    for (int i = 0; i < 2; ++i) {
      // A layout: lanes 0-15 K in {0..7,16..23}, lanes 16-31 +8
      const __bf16* p = &sA[(wm + i * 16 + l16) * LDS_STRIDE + halfl * 8];
      afrag[i] = make16(*(const v8bf*)p, *(const v8bf*)(p + 16));
    }
#pragma unroll
    for (int j = 0; j < 4; ++j) {
      // B layout: lanes 0-15 K in {0..7,16..23}, lanes 16-31 +8
      const __bf16* p = &sB[(wn + j * 16 + l16) * LDS_STRIDE + halfl * 8];
      bfrag[j] = make16(*(const v8bf*)p, *(const v8bf*)(p + 16));
    }
#pragma unroll
    for (int i = 0; i < 2; ++i)
#pragma unroll
      for (int j = 0; j < 4; ++j)
        acc[i][j] = wmma_bf16(afrag[i], bfrag[j], acc[i][j]);

    __syncthreads();
  }

  // ---- epilogue: C layout row M=(lane/16)*8+r, col N=lane&15 ----
  const int crow = halfl * 8;
#pragma unroll
  for (int j = 0; j < 4; ++j) {
    const int col = n0 + wn + j * 16 + l16;
    const float bj = bias[col];
#pragma unroll
    for (int i = 0; i < 2; ++i)
#pragma unroll
      for (int r = 0; r < 8; ++r) {
        int row = m0 + wm + i * 16 + crow + r;
        float v = acc[i][j][r] + bj;
        if (OUT_BF16)
          ((__bf16*)Oraw)[(size_t)row * N + col] = f2bf(v);
        else
          ((float*)Oraw)[(size_t)row * N + col] = v;
      }
  }
}

// ---------------------------------------------------------------------------
// Kernel 2: RoPE in place on bf16 qkv buffer (q and k halves).
// qkv layout: [B*T, 3072]; head h at cols h*64; pairs (2i, 2i+1).
// ---------------------------------------------------------------------------
__global__ void __launch_bounds__(256)
rope_kernel(__bf16* __restrict__ qkv) {
  int idx = blockIdx.x * blockDim.x + threadIdx.x;  // 2^21 total
  int i = idx & 31;
  int h = (idx >> 5) & 15;
  int t = (idx >> 9) & 2047;
  int b = idx >> 20;
  size_t base = ((size_t)(b * 2048 + t)) * 3072 + h * 64 + 2 * i;
  float freq = __powf(10000.0f, -(float)(2 * i) * (1.0f / 64.0f));
  float ang = (float)t * freq;
  float s, c;
  __sincosf(ang, &s, &c);
  {  // q
    float x1 = bf2f(qkv[base]), x2 = bf2f(qkv[base + 1]);
    qkv[base]     = f2bf(x1 * c - x2 * s);
    qkv[base + 1] = f2bf(x1 * s + x2 * c);
  }
  {  // k
    float x1 = bf2f(qkv[base + 1024]), x2 = bf2f(qkv[base + 1025]);
    qkv[base + 1024] = f2bf(x1 * c - x2 * s);
    qkv[base + 1025] = f2bf(x1 * s + x2 * c);
  }
}

// ---------------------------------------------------------------------------
// Kernel 3: causal flash attention, bf16 WMMA, online softmax.
// Grid: (T/64, B*H). Block: 128 threads (4 waves), wave w owns 16 query rows.
// ---------------------------------------------------------------------------
#define KSTRIDE 72   // sK row stride (64 data + 8 pad), 16B aligned
#define PSTRIDE 40   // sP/sVt row stride (32 data + 8 pad)

__global__ void __launch_bounds__(128)
flash_attn(const __bf16* __restrict__ qkv, __bf16* __restrict__ y) {
  constexpr int T = 2048, CH = 1024, ROW = 3072;

  __shared__ __bf16 sK[32 * KSTRIDE];       // [key][dim]
  __shared__ __bf16 sVt[64 * PSTRIDE];      // [dim][key] (transposed)
  __shared__ __bf16 sP[4][16 * PSTRIDE];    // per-wave P tile [row][key]

  const int tid  = threadIdx.x;
  const int wave = tid >> 5;
  const int lane = tid & 31;
  const int halfl = lane >> 4;
  const int l16   = lane & 15;

  const int b = blockIdx.y >> 4;
  const int h = blockIdx.y & 15;
  const int q0 = blockIdx.x * 64;
  const size_t baseQKV = (size_t)b * T * ROW;
  const size_t baseY   = (size_t)b * T * CH;

  // preload Q fragments (A layout): rows q0+wave*16 .. +15, dims 0..63
  v16bf qf[2];
  {
    const int qt = q0 + wave * 16 + l16;
    const __bf16* qp = qkv + baseQKV + (size_t)qt * ROW + h * 64;
#pragma unroll
    for (int c = 0; c < 2; ++c) {
      const __bf16* p = qp + c * 32 + halfl * 8;
      qf[c] = make16(*(const v8bf*)p, *(const v8bf*)(p + 16));
    }
  }

  v8f ao[4];
#pragma unroll
  for (int n = 0; n < 4; ++n)
#pragma unroll
    for (int r = 0; r < 8; ++r) ao[n][r] = 0.0f;
  float mrow[8], lsum[8];
#pragma unroll
  for (int r = 0; r < 8; ++r) { mrow[r] = -3.0e38f; lsum[r] = 0.0f; }

  const int kend = q0 + 64;  // keys [0, kend) in blocks of 32
  for (int kb = 0; kb < kend; kb += 32) {
    // ---- stage K tile and transposed V tile ----
    {
      const int row = tid >> 2;   // 0..31 (key within block)
      const int seg = tid & 3;    // 16-dim segment
      const __bf16* kp = qkv + baseQKV + (size_t)(kb + row) * ROW + CH + h * 64 + seg * 16;
      const __bf16* vp = qkv + baseQKV + (size_t)(kb + row) * ROW + 2 * CH + h * 64 + seg * 16;
      if (kb + 32 < kend) {  // prefetch next key block (global_prefetch_b8)
        __builtin_prefetch(kp + 32 * ROW, 0, 0);
        __builtin_prefetch(vp + 32 * ROW, 0, 0);
      }
      *(v8bf*)&sK[row * KSTRIDE + seg * 16]     = *(const v8bf*)kp;
      *(v8bf*)&sK[row * KSTRIDE + seg * 16 + 8] = *(const v8bf*)(kp + 8);
#pragma unroll
      for (int e = 0; e < 16; ++e) sVt[(seg * 16 + e) * PSTRIDE + row] = vp[e];
    }
    __syncthreads();

    // ---- S = Q K^T : two 16x16 key tiles, K-dim = d (2 chunks of 32) ----
    v8f s0, s1;
#pragma unroll
    for (int r = 0; r < 8; ++r) { s0[r] = 0.0f; s1[r] = 0.0f; }
#pragma unroll
    for (int c = 0; c < 2; ++c) {
      const __bf16* p0 = &sK[l16 * KSTRIDE + c * 32 + halfl * 8];
      s0 = wmma_bf16(qf[c], make16(*(const v8bf*)p0, *(const v8bf*)(p0 + 16)), s0);
      const __bf16* p1 = &sK[(16 + l16) * KSTRIDE + c * 32 + halfl * 8];
      s1 = wmma_bf16(qf[c], make16(*(const v8bf*)p1, *(const v8bf*)(p1 + 16)), s1);
    }

    // ---- causal mask (only near diagonal), online softmax, write P ----
    const int qbase = q0 + wave * 16 + halfl * 8;
    const bool needs_mask = (kb + 31 > q0 + wave * 16);  // wave-uniform
#pragma unroll
    for (int r = 0; r < 8; ++r) {
      const int qi = qbase + r;
      float a0, a1;
      if (needs_mask) {
        a0 = (kb + l16      <= qi) ? s0[r] * 0.125f : -3.0e38f;
        a1 = (kb + 16 + l16 <= qi) ? s1[r] * 0.125f : -3.0e38f;
      } else {
        a0 = s0[r] * 0.125f;
        a1 = s1[r] * 0.125f;
      }
      float mb = fmaxf(a0, a1);
      mb = fmaxf(mb, __shfl_xor(mb, 1, 32));
      mb = fmaxf(mb, __shfl_xor(mb, 2, 32));
      mb = fmaxf(mb, __shfl_xor(mb, 4, 32));
      mb = fmaxf(mb, __shfl_xor(mb, 8, 32));
      float mnew  = fmaxf(mrow[r], mb);
      float scale = __expf(mrow[r] - mnew);
      float e0 = __expf(a0 - mnew);
      float e1 = __expf(a1 - mnew);
      float rs = e0 + e1;
      rs += __shfl_xor(rs, 1, 32);
      rs += __shfl_xor(rs, 2, 32);
      rs += __shfl_xor(rs, 4, 32);
      rs += __shfl_xor(rs, 8, 32);
      lsum[r] = lsum[r] * scale + rs;
      mrow[r] = mnew;
#pragma unroll
      for (int n = 0; n < 4; ++n) ao[n][r] *= scale;
      const int prow = halfl * 8 + r;
      sP[wave][prow * PSTRIDE + l16]      = f2bf(e0);
      sP[wave][prow * PSTRIDE + 16 + l16] = f2bf(e1);
    }
    __syncthreads();  // P visible across lanes (and KV reads complete)

    // ---- O += P V : K-dim = 32 keys, 4 dim tiles of 16 ----
    v16bf pf;
    {
      const __bf16* p = &sP[wave][l16 * PSTRIDE + halfl * 8];
      pf = make16(*(const v8bf*)p, *(const v8bf*)(p + 16));
    }
#pragma unroll
    for (int n = 0; n < 4; ++n) {
      const __bf16* p = &sVt[(n * 16 + l16) * PSTRIDE + halfl * 8];
      v16bf vf = make16(*(const v8bf*)p, *(const v8bf*)(p + 16));
      ao[n] = wmma_bf16(pf, vf, ao[n]);
    }
    __syncthreads();  // done with sK/sVt before next stage
  }

  // ---- epilogue: normalize and store y (bf16) ----
#pragma unroll
  for (int n = 0; n < 4; ++n)
#pragma unroll
    for (int r = 0; r < 8; ++r) {
      int row = q0 + wave * 16 + halfl * 8 + r;
      int col = h * 64 + n * 16 + l16;
      y[baseY + (size_t)row * CH + col] = f2bf(ao[n][r] / lsum[r]);
    }
}

// ---------------------------------------------------------------------------
// Launcher
// ---------------------------------------------------------------------------
extern "C" void kernel_launch(void* const* d_in, const int* in_sizes, int n_in,
                              void* d_out, int out_size, void* d_ws, size_t ws_size,
                              hipStream_t stream) {
  (void)in_sizes; (void)n_in; (void)out_size; (void)ws_size;
  const float* x     = (const float*)d_in[0];
  const float* w_in  = (const float*)d_in[1];
  const float* b_in  = (const float*)d_in[2];
  const float* w_out = (const float*)d_in[3];
  const float* b_out = (const float*)d_in[4];

  __bf16* qkv = (__bf16*)d_ws;                       // [4096, 3072] bf16
  __bf16* y   = qkv + (size_t)4096 * 3072;           // [4096, 1024] bf16

  // 1) qkv = x @ w_in^T + b_in   (M=4096, N=3072, K=1024)
  dim3 g1(3072 / 128, 4096 / 128);
  wmma_gemm<false, true><<<g1, 256, 0, stream>>>(x, w_in, b_in, qkv, 4096, 3072, 1024);

  // 2) RoPE on q,k halves in place
  rope_kernel<<<(2 * 2048 * 16 * 32) / 256, 256, 0, stream>>>(qkv);

  // 3) causal flash attention -> y
  dim3 g3(2048 / 64, 2 * 16);
  flash_attn<<<g3, 128, 0, stream>>>(qkv, y);

  // 4) out = y @ w_out^T + b_out  (M=4096, N=1024, K=1024), fp32 out
  dim3 g4(1024 / 128, 4096 / 128);
  wmma_gemm<true, false><<<g4, 256, 0, stream>>>(y, w_out, b_out, d_out, 4096, 1024, 1024);
}
